// M_PCNN_50629074486111
// MI455X (gfx1250) — matrix operand order repaired and checked by reading, not compiled, not run on hardware
//
#include <hip/hip_runtime.h>
#include <stdint.h>

// ---------------------------------------------------------------------------
// MI455X persistent LSTM-ensemble kernel (round 3).
//   - bf16 WMMA (v_wmma_f32_16x16x32_bf16) for all GEMMs, f32 accumulate.
//   - Weights repacked per launch into WMMA B-fragment layout; streamed from
//     L2 every step. A single laundered zero *offset* (asm "+s") is added to
//     every global base pointer per timestep: blocks LICM from hoisting the
//     weight loads out of the 96-step loop (round 1: hoist -> 1024-VGPR spill)
//     while preserving pointer provenance so loads lower to global_load_b128
//     (round 2's pointer laundering degraded them to flat_load_b128, which
//     also ties up DScnt and needs 64-bit vector address adds).
//   - c0/c1 cell state lives in LDS f32; h0/h1 ping-pong in LDS as bf16.
//   - grid = 16 models x 4 batch chunks = 64 blocks; device-wide spin barrier
//     per timestep to exchange Tp (written straight into d_out).
// ---------------------------------------------------------------------------

#define RR   16
#define HH   128
#define NIN  7
#define NH1_ 64
#define NO1_ 64
#define NO2_ 32
#define BB   128
#define TT   96
#define FF   53
#define GG   512
#define BC   4            // batch chunks
#define MC   32           // batch rows per block
#define NBLK (RR * BC)    // 64 blocks
#define NTHR 128          // 4 wave32

typedef __attribute__((ext_vector_type(16))) __bf16 v16bf;
typedef __attribute__((ext_vector_type(8)))  float  v8f;

union B16F { v16bf v; uint4 q[2]; };

__device__ __forceinline__ unsigned short f2bfu(float f) {
  unsigned u = __float_as_uint(f);
  u += 0x7FFFu + ((u >> 16) & 1u);          // round-to-nearest-even
  return (unsigned short)(u >> 16);
}

__device__ __forceinline__ float sigf(float x) { return 1.0f / (1.0f + __expf(-x)); }

__device__ __forceinline__ v8f zero8() {
  v8f z;
#pragma unroll
  for (int i = 0; i < 8; ++i) z[i] = 0.0f;
  return z;
}

__device__ __forceinline__ v8f wmma_bf16(v16bf a, v16bf b, v8f c) {
  // (neg_a, A, neg_b, B, c_mod, C, reuse_a, reuse_b)
  return __builtin_amdgcn_wmma_f32_16x16x32_bf16(false, a, false, b, (short)0, c,
                                                 false, false);
}

// A fragment (16x32 bf16) from LDS row-major [rows][Kt].
// lane<16: m=lane,    k = kbase + {0..7, 16..23}
// lane>=16:m=lane-16, k = kbase + {8..15, 24..31}
__device__ __forceinline__ v16bf load_a(const unsigned short* S, int Kt,
                                        int mbase, int kbase) {
  int lane = threadIdx.x & 31;
  int m  = mbase + (lane & 15);
  int k0 = kbase + ((lane >> 4) << 3);
  B16F f;
  f.q[0] = *(const uint4*)(S + m * Kt + k0);
  f.q[1] = *(const uint4*)(S + m * Kt + k0 + 16);
  return f.v;
}

// B fragment (32x16 bf16) pre-packed in global: lane reads 32 contiguous bytes.
__device__ __forceinline__ v16bf load_b(const unsigned short* Gp) {
  int lane = threadIdx.x & 31;
  B16F f;
  const uint4* p = (const uint4*)(Gp + (lane << 4));
  f.q[0] = p[0];
  f.q[1] = p[1];
  return f.v;
}

// One LSTM cell layer for this block's 32 rows. Each wave owns (mtile, h-half);
// the four gate tiles (i,f,g,o) for one 16-wide hidden group are accumulated
// together so the recurrence update needs no cross-wave traffic. Cell state is
// read-modify-written in LDS (wave-private region, race-free).
__device__ __forceinline__ void lstm_cell(
    const unsigned short* __restrict__ Ain, int K1,     // LDS [32][K1] bf16
    const unsigned short* __restrict__ Ah,              // LDS [32][128] bf16
    const unsigned short* __restrict__ WihF,            // r-offset frag base
    const unsigned short* __restrict__ WhhF,            // r-offset frag base
    const float* __restrict__ bias,                     // [512] (r-offset)
    float* __restrict__ cL,                             // LDS [32][128] f32
    unsigned short* __restrict__ Hout,                  // LDS [32][128] bf16
    float* __restrict__ Hf32) {                         // optional f32 copy
  const int lane  = threadIdx.x & 31;
  const int wave  = threadIdx.x >> 5;
  const int mtile = wave >> 1;   // 0..1
  const int hg    = wave & 1;    // hidden half
  const int KS1   = K1 >> 5;
#pragma unroll 1
  for (int nh = 0; nh < 4; ++nh) {
    v8f acc[4];
#pragma unroll
    for (int q = 0; q < 4; ++q) acc[q] = zero8();
    for (int ks = 0; ks < KS1; ++ks) {
      v16bf a = load_a(Ain, K1, mtile * 16, ks * 32);
#pragma unroll
      for (int q = 0; q < 4; ++q) {
        int nt = q * 8 + hg * 4 + nh;
        acc[q] = wmma_bf16(a, load_b(WihF + ((nt * KS1 + ks) << 9)), acc[q]);
      }
    }
#pragma unroll
    for (int ks = 0; ks < 4; ++ks) {
      v16bf a = load_a(Ah, HH, mtile * 16, ks * 32);
#pragma unroll
      for (int q = 0; q < 4; ++q) {
        int nt = q * 8 + hg * 4 + nh;
        acc[q] = wmma_bf16(a, load_b(WhhF + ((nt * 4 + ks) << 9)), acc[q]);
      }
    }
    const int   ncol = (hg * 4 + nh) * 16 + (lane & 15);
    const float bi  = bias[          ncol];
    const float bf_ = bias[1 * HH  + ncol];
    const float bg_ = bias[2 * HH  + ncol];
    const float bo  = bias[3 * HH  + ncol];
    const int   mrow = mtile * 16 + ((lane >> 4) << 3);
#pragma unroll
    for (int v = 0; v < 8; ++v) {
      float iv = sigf(acc[0][v] + bi);
      float fv = sigf(acc[1][v] + bf_);
      float gv = tanhf(acc[2][v] + bg_);
      float ov = sigf(acc[3][v] + bo);
      float cv = cL[(mrow + v) * HH + ncol];
      float cn = fv * cv + iv * gv;
      cL[(mrow + v) * HH + ncol] = cn;
      float hv = ov * tanhf(cn);
      Hout[(mrow + v) * HH + ncol] = f2bfu(hv);
      if (Hf32) Hf32[(mrow + v) * HH + ncol] = hv;
    }
  }
}

// ---------------------------------------------------------------------------
// Prep: pack fp32 weights into bf16 WMMA B-fragments.
// dst[((r*NT+nt)*KS+ks)*512 + lane*16 + e] ; n = nt*16 + (lane&15),
// k = ks*32 + ((lane>>4)<<3) + (e&7) + ((e>>3)<<4).  transposed=0: W[r][n][k]; =1: W[r][k][n].
// ---------------------------------------------------------------------------
__global__ void pack_frags(const float* __restrict__ W,
                           unsigned short* __restrict__ dst,
                           int NT, int KS, int Kdim, int transposed) {
  const int  NTOT  = NT * 16;
  const long total = (long)RR * NT * KS * 512;
  for (long i = (long)blockIdx.x * blockDim.x + threadIdx.x; i < total;
       i += (long)gridDim.x * blockDim.x) {
    int  e    = (int)(i & 15);
    long j    = i >> 4;
    int  lane = (int)(j & 31);  j >>= 5;
    int  ks   = (int)(j % KS);  j /= KS;
    int  nt   = (int)(j % NT);
    int  r    = (int)(j / NT);
    int  n    = nt * 16 + (lane & 15);
    int  k    = ks * 32 + ((lane >> 4) << 3) + (e & 7) + ((e >> 3) << 4);
    float v = 0.0f;
    if (k < Kdim)
      v = transposed ? W[((long)r * Kdim + k) * NTOT + n]
                     : W[((long)r * NTOT + n) * Kdim + k];
    dst[i] = f2bfu(v);
  }
}

// ---------------------------------------------------------------------------
// Dynamic-LDS layout (bytes)
// ---------------------------------------------------------------------------
#define SM_TP    0        // float[32*16]
#define SM_FEAT  2048     // float[32*8]
#define SM_LDE   3072     // float[32*2]
#define SM_CW    3328     // float[24]
#define SM_SCAL  3424     // float[4]
#define SM_RED   3456     // float[32*8]
#define SM_STAT  4480     // float[32*2]
#define SM_Z     4736     // ushort[32*64]
#define SM_H0    8832     // ushort[2*32*128]
#define SM_H1    25216    // ushort[2*32*128]
#define SM_HF    41600    // float[32*128]
#define SM_Y     57984    // ushort[32*128]
#define SM_Y2    66176    // ushort[32*64]
#define SM_Y3    70272    // float[32*32]
#define SM_C0    74368    // float[32*128]
#define SM_C1    90752    // float[32*128]
#define SMEM_BYTES 107136

__global__ void __launch_bounds__(NTHR, 1)
rnn_persistent(const float* __restrict__ x,
               const float* __restrict__ in_W,  const float* __restrict__ in_b,
               const float* __restrict__ b0g,   const float* __restrict__ b1g,
               const float* __restrict__ norm_g,const float* __restrict__ norm_b,
               const float* __restrict__ ob1,   const float* __restrict__ ob2,
               const float* __restrict__ oW3,   const float* __restrict__ ob3,
               const float* __restrict__ log_a, const float* __restrict__ log_b,
               const float* __restrict__ log_c, const float* __restrict__ log_d,
               const float* __restrict__ init_h,const float* __restrict__ init_c,
               const unsigned short* __restrict__ wih0F,
               const unsigned short* __restrict__ whh0F,
               const unsigned short* __restrict__ wih1F,
               const unsigned short* __restrict__ whh1F,
               const unsigned short* __restrict__ ow1F,
               const unsigned short* __restrict__ ow2F,
               unsigned* __restrict__ syncCnt,
               float* __restrict__ out) {
  extern __shared__ char smem[];
  float*          TpS    = (float*)(smem + SM_TP);
  float*          featsS = (float*)(smem + SM_FEAT);
  float*          lDE    = (float*)(smem + SM_LDE);
  float*          cwS    = (float*)(smem + SM_CW);
  float*          scalS  = (float*)(smem + SM_SCAL);
  float*          red    = (float*)(smem + SM_RED);
  float*          stats  = (float*)(smem + SM_STAT);
  unsigned short* zS     = (unsigned short*)(smem + SM_Z);
  unsigned short* h0S    = (unsigned short*)(smem + SM_H0);
  unsigned short* h1S    = (unsigned short*)(smem + SM_H1);
  float*          Hf     = (float*)(smem + SM_HF);
  unsigned short* yS     = (unsigned short*)(smem + SM_Y);
  unsigned short* y2S    = (unsigned short*)(smem + SM_Y2);
  float*          y3f    = (float*)(smem + SM_Y3);
  float*          c0L    = (float*)(smem + SM_C0);
  float*          c1L    = (float*)(smem + SM_C1);

  const int r   = blockIdx.x & (RR - 1);
  const int bc  = blockIdx.x >> 4;
  const int bg0 = bc * MC;
  const int tid = threadIdx.x;
  const int lane = tid & 31;
  const int wave = tid >> 5;

  const unsigned short* wih0r = wih0F + (size_t)r * 32 * 2 * 512;
  const unsigned short* whh0r = whh0F + (size_t)r * 32 * 4 * 512;
  const unsigned short* wih1r = wih1F + (size_t)r * 32 * 4 * 512;
  const unsigned short* whh1r = whh1F + (size_t)r * 32 * 4 * 512;
  const unsigned short* ow1r  = ow1F  + (size_t)r * 4 * 4 * 512;
  const unsigned short* ow2r  = ow2F  + (size_t)r * 2 * 2 * 512;

  // ---- init persistent LDS state ----
  for (int e = tid; e < MC * HH; e += NTHR) {
    int h = e & (HH - 1);
    h0S[e] = f2bfu(init_h[(r * 2 + 0) * HH + h]);
    h1S[e] = f2bfu(init_h[(r * 2 + 1) * HH + h]);
    c0L[e] = init_c[(r * 2 + 0) * HH + h];
    c1L[e] = init_c[(r * 2 + 1) * HH + h];
  }
  if (tid < MC) {
    lDE[tid * 2 + 0] = x[((size_t)(bg0 + tid) * TT) * FF + r];
    lDE[tid * 2 + 1] = 0.0f;
  }
  if (tid < 24) cwS[tid] = __expf(log_c[tid]);
  if (tid == 24) scalS[0] = __expf(log_a[r]);
  if (tid == 25) scalS[1] = __expf(log_b[r]);
  if (tid == 26) scalS[2] = __expf(log_d[r]);
  __syncthreads();

  for (int t = 0; t < TT; ++t) {
    const int cur = (t & 1) * MC * HH;
    const int nxt = ((t + 1) & 1) * MC * HH;

    // Launder a single zero offset (SGPR) per timestep and add it to every
    // global base pointer: the addresses become loop-variant (no LICM hoist
    // of weight loads -> no spill) while pointer provenance is preserved so
    // loads lower to global_load_b128 saddr form (not flat_load).
    size_t zoff = 0;
    asm volatile("" : "+s"(zoff));
    const float* xp   = x      + zoff;
    const float* inWp = in_W   + zoff;
    const float* inBp = in_b   + zoff;
    const float* b0p  = b0g + r * GG + zoff;
    const float* b1p  = b1g + r * GG + zoff;
    const float* ngp  = norm_g + zoff;
    const float* nbp  = norm_b + zoff;
    const float* ob1p = ob1    + zoff;
    const float* ob2p = ob2    + zoff;
    const float* oW3p = oW3    + zoff;
    const float* ob3p = ob3    + zoff;
    float*       outp = out    + zoff;
    const unsigned short* wih0p = wih0r + zoff;
    const unsigned short* whh0p = whh0r + zoff;
    const unsigned short* wih1p = wih1r + zoff;
    const unsigned short* whh1p = whh1r + zoff;
    const unsigned short* ow1p  = ow1r  + zoff;
    const unsigned short* ow2p  = ow2r  + zoff;

    // ---- Stage A: gather Tp (cross-model coupling) + build features ----
    if (tid < MC) {
      int bg = bg0 + tid;
      if (t == 0) {
        for (int n = 0; n < RR; ++n)
          TpS[tid * RR + n] = xp[((size_t)bg * TT) * FF + n];
      } else {
        for (int n = 0; n < RR; ++n)
          TpS[tid * RR + n] = outp[((size_t)bg * TT + (t - 1)) * RR + n];
      }
      const float* xr = xp + ((size_t)bg * TT + t) * FF;
      featsS[tid * 8 + 0] = TpS[tid * RR + r];
      featsS[tid * 8 + 1] = xr[RR + r];
      featsS[tid * 8 + 2] = xr[48];
      featsS[tid * 8 + 3] = xr[49];
      featsS[tid * 8 + 4] = xr[50];
      featsS[tid * 8 + 5] = xr[51];
      featsS[tid * 8 + 6] = xr[52];
    }
    __syncthreads();

    // ---- Stage B: z = relu(feats @ in_W + in_b), K=7 (scalar, tiny) ----
    {
      int row = tid >> 2, seg = tid & 3;
      float f0 = featsS[row * 8 + 0], f1 = featsS[row * 8 + 1];
      float f2 = featsS[row * 8 + 2], f3 = featsS[row * 8 + 3];
      float f4 = featsS[row * 8 + 4], f5 = featsS[row * 8 + 5];
      float f6 = featsS[row * 8 + 6];
      for (int oo = seg * 16; oo < seg * 16 + 16; ++oo) {
        float acc = inBp[r * NH1_ + oo];
        acc += f0 * inWp[((r * NIN + 0) * NH1_) + oo];
        acc += f1 * inWp[((r * NIN + 1) * NH1_) + oo];
        acc += f2 * inWp[((r * NIN + 2) * NH1_) + oo];
        acc += f3 * inWp[((r * NIN + 3) * NH1_) + oo];
        acc += f4 * inWp[((r * NIN + 4) * NH1_) + oo];
        acc += f5 * inWp[((r * NIN + 5) * NH1_) + oo];
        acc += f6 * inWp[((r * NIN + 6) * NH1_) + oo];
        zS[row * NH1_ + oo] = f2bfu(fmaxf(acc, 0.0f));
      }
    }
    __syncthreads();

    // ---- LSTM cells (WMMA) ----
    lstm_cell(zS, NH1_, h0S + cur, wih0p, whh0p, b0p, c0L, h0S + nxt,
              (float*)0);
    __syncthreads();
    lstm_cell(h0S + nxt, HH, h1S + cur, wih1p, whh1p, b1p, c1L, h1S + nxt,
              Hf);
    __syncthreads();

    // ---- LayerNorm over H=128 ----
    {
      int row = tid >> 2, q = tid & 3;
      float s = 0.f, ss = 0.f;
      for (int c2 = q * 32; c2 < q * 32 + 32; ++c2) {
        float v = Hf[row * HH + c2];
        s += v; ss += v * v;
      }
      red[(row * 4 + q) * 2 + 0] = s;
      red[(row * 4 + q) * 2 + 1] = ss;
    }
    __syncthreads();
    if (tid < MC) {
      float s = 0.f, ss = 0.f;
      for (int q = 0; q < 4; ++q) {
        s  += red[(tid * 4 + q) * 2 + 0];
        ss += red[(tid * 4 + q) * 2 + 1];
      }
      float mu  = s * (1.0f / HH);
      float var = ss * (1.0f / HH) - mu * mu;
      stats[tid * 2 + 0] = mu;
      stats[tid * 2 + 1] = rsqrtf(var + 1e-5f);
    }
    __syncthreads();
    {
      int row = tid >> 2, q = tid & 3;
      float mu = stats[row * 2 + 0], rv = stats[row * 2 + 1];
      for (int c2 = q * 32; c2 < q * 32 + 32; ++c2) {
        float v = (Hf[row * HH + c2] - mu) * rv * ngp[r * HH + c2] +
                  nbp[r * HH + c2];
        yS[row * HH + c2] = f2bfu(v);
      }
    }
    __syncthreads();

    // ---- MLP1: [32x128] @ [128x64], tanh (WMMA) ----
#pragma unroll 1
    for (int s2 = 0; s2 < 2; ++s2) {
      int tt2 = wave * 2 + s2;
      int mt = tt2 >> 2, nt = tt2 & 3;
      v8f acc = zero8();
#pragma unroll
      for (int ks = 0; ks < 4; ++ks)
        acc = wmma_bf16(load_a(yS, HH, mt * 16, ks * 32),
                        load_b(ow1p + ((nt * 4 + ks) << 9)), acc);
      int col  = nt * 16 + (lane & 15);
      float bb = ob1p[r * NO1_ + col];
      int mrow = mt * 16 + ((lane >> 4) << 3);
#pragma unroll
      for (int v = 0; v < 8; ++v)
        y2S[(mrow + v) * NO1_ + col] = f2bfu(tanhf(acc[v] + bb));
    }
    __syncthreads();

    // ---- MLP2: [32x64] @ [64x32], tanh (WMMA) ----
    {
      int mt = wave >> 1, nt = wave & 1;
      v8f acc = zero8();
#pragma unroll
      for (int ks = 0; ks < 2; ++ks)
        acc = wmma_bf16(load_a(y2S, NO1_, mt * 16, ks * 32),
                        load_b(ow2p + ((nt * 2 + ks) << 9)), acc);
      int col  = nt * 16 + (lane & 15);
      float bb = ob2p[r * NO2_ + col];
      int mrow = mt * 16 + ((lane >> 4) << 3);
#pragma unroll
      for (int v = 0; v < 8; ++v)
        y3f[(mrow + v) * NO2_ + col] = tanhf(acc[v] + bb);
    }
    __syncthreads();

    // ---- Head + thermal dynamics (scalar, 32 rows) ----
    if (tid < MC) {
      float acc = ob3p[r];
      for (int j = 0; j < NO2_; ++j)
        acc += y3f[tid * NO2_ + j] * oW3p[r * NO2_ + j];
      float d  = tanhf(acc);
      float lD = lDE[tid * 2 + 0], lE = lDE[tid * 2 + 1];
      float D  = lD + d * 0.1f;                               // /DIV
      int bg = bg0 + tid;
      const float* xr = xp + ((size_t)bg * TT + t) * FF;
      float power = xr[RR + r];
      float cse   = xr[2 * RR + r];
      float tout  = xr[3 * RR];
      float dp = power - 0.5f;                                // ZERO_POWER
      float aw = scalS[0], bw = scalS[1], dw = scalS[2];
      float E = lE + (cse > 0.5f ? aw * 0.01f * dp : dw * 0.01f * dp);
      float Tpo = TpS[tid * RR + r];
      E -= bw * 0.005f * ((Tpo - tout) * (20.0f / 0.8f));
      const int NB1a[24] = {0,1,2,4,5,6,8,9,10,12,13,14,0,1,2,3,4,5,6,7,8,9,10,11};
      const int NB2a[24] = {1,2,3,5,6,7,9,10,11,13,14,15,4,5,6,7,8,9,10,11,12,13,14,15};
#pragma unroll
      for (int k = 0; k < 24; ++k) {
        int n1 = NB1a[k], n2 = NB2a[k];
        if (n1 == r || n2 == r) {
          float flux = cwS[k] *
              ((TpS[tid * RR + n1] - TpS[tid * RR + n2]) * (20.0f / 0.8f));
          if (n1 == r) E -= 0.005f * flux;
          if (n2 == r) E += 0.005f * flux;
        }
      }
      lDE[tid * 2 + 0] = D;
      lDE[tid * 2 + 1] = E;
      outp[((size_t)bg * TT + t) * RR + r] = D + E;  // == next step's Tp col
    }

    // ---- device-wide barrier (monotonic counter) ----
    __syncthreads();
    if (tid == 0) {
      __threadfence();
      atomicAdd(syncCnt, 1u);
      unsigned target = (unsigned)(t + 1) * gridDim.x;
      while (*(volatile unsigned*)syncCnt < target) {
        __builtin_amdgcn_s_sleep(1);
      }
      __threadfence();
    }
    __syncthreads();
  }
}

// ---------------------------------------------------------------------------
extern "C" void kernel_launch(void* const* d_in, const int* in_sizes, int n_in,
                              void* d_out, int out_size, void* d_ws,
                              size_t ws_size, hipStream_t stream) {
  (void)in_sizes; (void)n_in; (void)out_size; (void)ws_size;
  const float* x      = (const float*)d_in[0];
  const float* in_W   = (const float*)d_in[1];
  const float* in_b   = (const float*)d_in[2];
  const float* Wih0   = (const float*)d_in[3];
  const float* Whh0   = (const float*)d_in[4];
  const float* b0     = (const float*)d_in[5];
  const float* Wih1   = (const float*)d_in[6];
  const float* Whh1   = (const float*)d_in[7];
  const float* b1     = (const float*)d_in[8];
  const float* ngam   = (const float*)d_in[9];
  const float* nbet   = (const float*)d_in[10];
  const float* oW1    = (const float*)d_in[11];
  const float* ob1    = (const float*)d_in[12];
  const float* oW2    = (const float*)d_in[13];
  const float* ob2    = (const float*)d_in[14];
  const float* oW3    = (const float*)d_in[15];
  const float* ob3    = (const float*)d_in[16];
  const float* log_a  = (const float*)d_in[17];
  const float* log_b  = (const float*)d_in[18];
  const float* log_c  = (const float*)d_in[19];
  const float* log_d  = (const float*)d_in[20];
  const float* init_h = (const float*)d_in[21];
  const float* init_c = (const float*)d_in[22];

  char* ws = (char*)d_ws;
  const size_t OFF_CNT  = 0;
  const size_t OFF_WIH0 = 256;
  const size_t SZ_WIH0  = (size_t)RR * 32 * 2 * 512 * 2;   // 1 MB
  const size_t SZ_G4    = (size_t)RR * 32 * 4 * 512 * 2;   // 2 MB
  const size_t OFF_WHH0 = OFF_WIH0 + SZ_WIH0;
  const size_t OFF_WIH1 = OFF_WHH0 + SZ_G4;
  const size_t OFF_WHH1 = OFF_WIH1 + SZ_G4;
  const size_t OFF_OW1  = OFF_WHH1 + SZ_G4;
  const size_t SZ_OW1   = (size_t)RR * 4 * 4 * 512 * 2;    // 256 KB
  const size_t OFF_OW2  = OFF_OW1 + SZ_OW1;

  unsigned short* wih0F = (unsigned short*)(ws + OFF_WIH0);
  unsigned short* whh0F = (unsigned short*)(ws + OFF_WHH0);
  unsigned short* wih1F = (unsigned short*)(ws + OFF_WIH1);
  unsigned short* whh1F = (unsigned short*)(ws + OFF_WHH1);
  unsigned short* ow1F  = (unsigned short*)(ws + OFF_OW1);
  unsigned short* ow2F  = (unsigned short*)(ws + OFF_OW2);
  unsigned*       cnt   = (unsigned*)(ws + OFF_CNT);

  hipMemsetAsync(cnt, 0, 256, stream);

  // Pack weights into WMMA B-fragment layout (bf16). Re-run every launch for
  // determinism (d_ws is poisoned once and never restored by the harness).
  pack_frags<<<1024, 256, 0, stream>>>(Wih0, wih0F, 32, 2, 64,  0);
  pack_frags<<<1024, 256, 0, stream>>>(Whh0, whh0F, 32, 4, 128, 0);
  pack_frags<<<1024, 256, 0, stream>>>(Wih1, wih1F, 32, 4, 128, 0);
  pack_frags<<<1024, 256, 0, stream>>>(Whh1, whh1F, 32, 4, 128, 0);
  pack_frags<<<256,  256, 0, stream>>>(oW1,  ow1F,  4,  4, 128, 1);
  pack_frags<<<64,   256, 0, stream>>>(oW2,  ow2F,  2,  2, 64,  1);

  rnn_persistent<<<dim3(NBLK), dim3(NTHR), SMEM_BYTES, stream>>>(
      x, in_W, in_b, b0, b1, ngam, nbet, ob1, ob2, oW3, ob3,
      log_a, log_b, log_c, log_d, init_h, init_c,
      wih0F, whh0F, wih1F, whh1F, ow1F, ow2F, cnt, (float*)d_out);
}
